// GAT_68865505624225
// MI455X (gfx1250) — matrix operand order repaired and checked by reading, not compiled, not run on hardware
//
#include <hip/hip_runtime.h>
#include <hip/hip_bf16.h>
#include <math.h>

typedef __attribute__((ext_vector_type(16))) __bf16 v16bf;
typedef __attribute__((ext_vector_type(8)))  float  v8f;

#define N_NODES   50000
#define F_IN      256
#define HIDC      64
#define HEADS     8
#define HH        512          // HEADS*HID
#define NCLSV     16
#define N_EDGE    400000
#define N_ET      450000       // edges + self loops
#define ROW_TILES 3125         // N_NODES/16
#define NEG_SLOPE 0.2f

__device__ __forceinline__ float lrelu(float a) { return a > 0.f ? a : NEG_SLOPE * a; }

// float atomic max via int/uint trick (valid for non-NaN, init -inf)
__device__ __forceinline__ void atomicMaxF(float* addr, float v) {
    if (v >= 0.f) atomicMax((int*)addr, __float_as_int(v));
    else          atomicMin((unsigned int*)addr, __float_as_uint(v));
}

__device__ __forceinline__ void edge_sd(const int* __restrict__ ei, int e, int& s, int& d) {
    if (e < N_EDGE) { s = ei[e]; d = ei[N_EDGE + e]; }
    else            { s = d = e - N_EDGE; }
}

// ---------------------------------------------------------------- fills
__global__ void fill_kernel(float* __restrict__ p, float v, int n) {
    int i = blockIdx.x * blockDim.x + threadIdx.x;
    if (i < n) p[i] = v;
}

// ---------------------------------------------------------------- W packing into WMMA B-fragment layout
// B (KxN) tile 32x16: lane L holds N=L%16, K = 16*(L>>4) + j (j=0..15), contiguous 32B per lane.
__global__ void pack_w1(const float* __restrict__ W1, __bf16* __restrict__ P) {
    int lane = threadIdx.x;          // 32
    int nt   = blockIdx.x;           // 0..31  (512/16 col tiles)
    int kt   = blockIdx.y;           // 0..7   (256/32 k tiles)
    int n    = nt * 16 + (lane & 15);
    int kb   = kt * 32 + 16 * (lane >> 4);
    v16bf t;
#pragma unroll
    for (int j = 0; j < 16; ++j) t[j] = (__bf16)W1[(kb + j) * HH + n];
    *(v16bf*)(P + (((size_t)nt * 8 + kt) * 32 + lane) * 16) = t;
}

__global__ void pack_w2(const float* __restrict__ W2, __bf16* __restrict__ P) {
    int lane = threadIdx.x;          // 32
    int kt   = blockIdx.x;           // 0..15 (512/32 k tiles), single col tile
    int n    = lane & 15;
    int kb   = kt * 32 + 16 * (lane >> 4);
    v16bf t;
#pragma unroll
    for (int j = 0; j < 16; ++j) t[j] = (__bf16)W2[(kb + j) * NCLSV + n];
    *(v16bf*)(P + (((size_t)kt) * 32 + lane) * 16) = t;
}

// ---------------------------------------------------------------- GEMM1: h = x @ W1   (N x 256) x (256 x 512)
// wave computes a 16x64 strip: 4 accumulators, A fragment reused 4x.
__global__ void __launch_bounds__(128) gemm1_kernel(const float* __restrict__ x,
                                                    const __bf16* __restrict__ P,
                                                    float* __restrict__ h) {
    int lane = threadIdx.x & 31;
    int wave = threadIdx.x >> 5;
    int rowTile = blockIdx.x * 4 + wave;
    if (rowTile >= ROW_TILES) return;
    int colGroup = blockIdx.y;                 // 0..7, 64 cols each
    int hi  = lane >> 4;
    int row = rowTile * 16 + (lane & 15);
    const float* xr = x + (size_t)row * F_IN;

    v8f acc[4] = {};
    for (int kt = 0; kt < 8; ++kt) {
        const float4* xp = (const float4*)(xr + kt * 32 + 16 * hi);
        float4 f0 = xp[0], f1 = xp[1];
        const float4* xq = (const float4*)(xr + kt * 32 + 16 * hi + 8);
        float4 f2 = xq[0], f3 = xq[1];
        v16bf a;
        a[0]=(__bf16)f0.x; a[1]=(__bf16)f0.y; a[2]=(__bf16)f0.z; a[3]=(__bf16)f0.w;
        a[4]=(__bf16)f1.x; a[5]=(__bf16)f1.y; a[6]=(__bf16)f1.z; a[7]=(__bf16)f1.w;
        a[8]=(__bf16)f2.x; a[9]=(__bf16)f2.y; a[10]=(__bf16)f2.z; a[11]=(__bf16)f2.w;
        a[12]=(__bf16)f3.x; a[13]=(__bf16)f3.y; a[14]=(__bf16)f3.z; a[15]=(__bf16)f3.w;
#pragma unroll
        for (int t = 0; t < 4; ++t) {
            int nt = colGroup * 4 + t;
            v16bf b = *(const v16bf*)(P + (((size_t)nt * 8 + kt) * 32 + lane) * 16);
            acc[t] = __builtin_amdgcn_wmma_f32_16x16x32_bf16(false, a, false, b,
                                                             (short)0, acc[t], false, false);
        }
    }
#pragma unroll
    for (int t = 0; t < 4; ++t) {
        int col = colGroup * 64 + t * 16 + (lane & 15);
        int rb  = rowTile * 16 + 8 * hi;
#pragma unroll
        for (int v = 0; v < 8; ++v)
            h[(size_t)(rb + v) * HH + col] = acc[t][v];
    }
}

// ---------------------------------------------------------------- GEMM2: h2 = helu @ W2  (N x 512) x (512 x 16)
__global__ void __launch_bounds__(128) gemm2_kernel(const float* __restrict__ xin,
                                                    const __bf16* __restrict__ P,
                                                    float* __restrict__ h2) {
    int lane = threadIdx.x & 31;
    int wave = threadIdx.x >> 5;
    int rowTile = blockIdx.x * 4 + wave;
    if (rowTile >= ROW_TILES) return;
    int hi  = lane >> 4;
    int row = rowTile * 16 + (lane & 15);
    const float* xr = xin + (size_t)row * HH;

    v8f acc = {};
    for (int kt = 0; kt < 16; ++kt) {
        const float4* xp = (const float4*)(xr + kt * 32 + 16 * hi);
        float4 f0 = xp[0], f1 = xp[1];
        const float4* xq = (const float4*)(xr + kt * 32 + 16 * hi + 8);
        float4 f2 = xq[0], f3 = xq[1];
        v16bf a;
        a[0]=(__bf16)f0.x; a[1]=(__bf16)f0.y; a[2]=(__bf16)f0.z; a[3]=(__bf16)f0.w;
        a[4]=(__bf16)f1.x; a[5]=(__bf16)f1.y; a[6]=(__bf16)f1.z; a[7]=(__bf16)f1.w;
        a[8]=(__bf16)f2.x; a[9]=(__bf16)f2.y; a[10]=(__bf16)f2.z; a[11]=(__bf16)f2.w;
        a[12]=(__bf16)f3.x; a[13]=(__bf16)f3.y; a[14]=(__bf16)f3.z; a[15]=(__bf16)f3.w;
        v16bf b = *(const v16bf*)(P + (((size_t)kt) * 32 + lane) * 16);
        acc = __builtin_amdgcn_wmma_f32_16x16x32_bf16(false, a, false, b,
                                                      (short)0, acc, false, false);
    }
    int col = lane & 15;
    int rb  = rowTile * 16 + 8 * hi;
#pragma unroll
    for (int v = 0; v < 8; ++v)
        h2[(size_t)(rb + v) * NCLSV + col] = acc[v];
}

// ---------------------------------------------------------------- attention logits (per-node halves)
__global__ void attn_logits1(const float* __restrict__ h, const float* __restrict__ asrc,
                             const float* __restrict__ adst,
                             float* __restrict__ als, float* __restrict__ ald) {
    int idx = blockIdx.x * blockDim.x + threadIdx.x;
    if (idx >= N_NODES * HEADS) return;
    int n = idx >> 3, hd = idx & 7;
    const float* hp = h + (size_t)n * HH + hd * HIDC;
    const float* as = asrc + hd * HIDC;
    const float* ad = adst + hd * HIDC;
    float s = 0.f, d = 0.f;
#pragma unroll 8
    for (int c = 0; c < HIDC; ++c) { float v = hp[c]; s += v * as[c]; d += v * ad[c]; }
    als[idx] = s; ald[idx] = d;
}

__global__ void attn_logits2(const float* __restrict__ h2, const float* __restrict__ asrc,
                             const float* __restrict__ adst,
                             float* __restrict__ als, float* __restrict__ ald) {
    int n = blockIdx.x * blockDim.x + threadIdx.x;
    if (n >= N_NODES) return;
    const float* hp = h2 + (size_t)n * NCLSV;
    float s = 0.f, d = 0.f;
#pragma unroll
    for (int c = 0; c < NCLSV; ++c) { float v = hp[c]; s += v * asrc[c]; d += v * adst[c]; }
    als[n] = s; ald[n] = d;
}

// ---------------------------------------------------------------- layer-1 edge softmax passes
__global__ void e1_max(const int* __restrict__ ei, const float* __restrict__ als,
                       const float* __restrict__ ald, float* __restrict__ m1) {
    int idx = blockIdx.x * blockDim.x + threadIdx.x;
    if (idx >= N_ET * HEADS) return;
    int e = idx >> 3, hd = idx & 7, s, d;
    edge_sd(ei, e, s, d);
    float a = lrelu(als[s * 8 + hd] + ald[d * 8 + hd]);
    atomicMaxF(&m1[d * 8 + hd], a);
}

__global__ void e1_sum(const int* __restrict__ ei, const float* __restrict__ als,
                       const float* __restrict__ ald, const float* __restrict__ m1,
                       float* __restrict__ den1) {
    int idx = blockIdx.x * blockDim.x + threadIdx.x;
    if (idx >= N_ET * HEADS) return;
    int e = idx >> 3, hd = idx & 7, s, d;
    edge_sd(ei, e, s, d);
    float a = lrelu(als[s * 8 + hd] + ald[d * 8 + hd]);
    atomicAdd(&den1[d * 8 + hd], expf(a - m1[d * 8 + hd]));
}

__global__ void e1_msg(const int* __restrict__ ei, const float* __restrict__ als,
                       const float* __restrict__ ald, const float* __restrict__ m1,
                       const float* __restrict__ den1, const float* __restrict__ h,
                       float* __restrict__ out1) {
    __shared__ float coef[HEADS];
    int e = blockIdx.x, s, d;
    edge_sd(ei, e, s, d);
    if (threadIdx.x < HEADS) {
        int hd = threadIdx.x;
        float a = lrelu(als[s * 8 + hd] + ald[d * 8 + hd]);
        coef[hd] = expf(a - m1[d * 8 + hd]) / den1[d * 8 + hd];
    }
    __syncthreads();
    const float* hs = h + (size_t)s * HH;
    float* od = out1 + (size_t)d * HH;
    for (int c = threadIdx.x; c < HH; c += 256)
        atomicAdd(&od[c], hs[c] * coef[c >> 6]);
}

// out1 = elu(out1 + b1)   (becomes layer-2 input)
__global__ void elu_bias(float* __restrict__ out1, const float* __restrict__ b1) {
    size_t idx = (size_t)blockIdx.x * blockDim.x + threadIdx.x;
    if (idx >= (size_t)N_NODES * HH) return;
    float v = out1[idx] + b1[idx & (HH - 1)];
    out1[idx] = v > 0.f ? v : expf(v) - 1.f;
}

// ---------------------------------------------------------------- layer-2 edge softmax passes
__global__ void e2_max(const int* __restrict__ ei, const float* __restrict__ als,
                       const float* __restrict__ ald, float* __restrict__ m2) {
    int e = blockIdx.x * blockDim.x + threadIdx.x;
    if (e >= N_ET) return;
    int s, d; edge_sd(ei, e, s, d);
    atomicMaxF(&m2[d], lrelu(als[s] + ald[d]));
}

__global__ void e2_sum(const int* __restrict__ ei, const float* __restrict__ als,
                       const float* __restrict__ ald, const float* __restrict__ m2,
                       float* __restrict__ den2) {
    int e = blockIdx.x * blockDim.x + threadIdx.x;
    if (e >= N_ET) return;
    int s, d; edge_sd(ei, e, s, d);
    atomicAdd(&den2[d], expf(lrelu(als[s] + ald[d]) - m2[d]));
}

__global__ void e2_msg(const int* __restrict__ ei, const float* __restrict__ als,
                       const float* __restrict__ ald, const float* __restrict__ m2,
                       const float* __restrict__ den2, const float* __restrict__ h2,
                       float* __restrict__ out) {
    int idx = blockIdx.x * blockDim.x + threadIdx.x;
    if (idx >= N_ET * NCLSV) return;
    int e = idx >> 4, c = idx & 15, s, d;
    edge_sd(ei, e, s, d);
    float a = lrelu(als[s] + ald[d]);
    float coef = expf(a - m2[d]) / den2[d];
    atomicAdd(&out[(size_t)d * NCLSV + c], h2[(size_t)s * NCLSV + c] * coef);
}

__global__ void bias2_kernel(float* __restrict__ out, const float* __restrict__ b2) {
    int idx = blockIdx.x * blockDim.x + threadIdx.x;
    if (idx >= N_NODES * NCLSV) return;
    out[idx] += b2[idx & 15];
}

// ----------------------------------------------------------------
extern "C" void kernel_launch(void* const* d_in, const int* in_sizes, int n_in,
                              void* d_out, int out_size, void* d_ws, size_t ws_size,
                              hipStream_t stream) {
    const float* x   = (const float*)d_in[0];
    const int*   ei  = (const int*)d_in[1];
    const float* W1  = (const float*)d_in[2];
    const float* as1 = (const float*)d_in[3];
    const float* ad1 = (const float*)d_in[4];
    const float* b1  = (const float*)d_in[5];
    const float* W2  = (const float*)d_in[6];
    const float* as2 = (const float*)d_in[7];
    const float* ad2 = (const float*)d_in[8];
    const float* b2  = (const float*)d_in[9];
    float* out = (float*)d_out;

    float* W = (float*)d_ws;
    size_t off = 0;
    float* h     = W + off; off += (size_t)N_NODES * HH;      // 25.6M
    float* out1  = W + off; off += (size_t)N_NODES * HH;      // 25.6M
    float* als1  = W + off; off += N_NODES * HEADS;
    float* ald1  = W + off; off += N_NODES * HEADS;
    float* m1    = W + off; off += N_NODES * HEADS;
    float* den1  = W + off; off += N_NODES * HEADS;
    float* als2v = W + off; off += N_NODES;
    float* ald2v = W + off; off += N_NODES;
    float* m2    = W + off; off += N_NODES;
    float* den2  = W + off; off += N_NODES;
    __bf16* pW1  = (__bf16*)(W + off); off += (F_IN * HH) / 2;   // 131072 bf16
    __bf16* pW2  = (__bf16*)(W + off); off += (HH * NCLSV) / 2;  // 8192 bf16
    float* h2 = h;  // reuse: h dead after e1_msg

    const float NEG_INF = -INFINITY;

    // pack weights to bf16 fragment layout
    pack_w1<<<dim3(32, 8), 32, 0, stream>>>(W1, pW1);
    pack_w2<<<16, 32, 0, stream>>>(W2, pW2);

    // init accumulators
    fill_kernel<<<(N_NODES * HEADS + 255) / 256, 256, 0, stream>>>(m1, NEG_INF, N_NODES * HEADS);
    fill_kernel<<<(N_NODES * HEADS + 255) / 256, 256, 0, stream>>>(den1, 0.f, N_NODES * HEADS);
    fill_kernel<<<(N_NODES * HH + 255) / 256, 256, 0, stream>>>(out1, 0.f, N_NODES * HH);
    fill_kernel<<<(N_NODES + 255) / 256, 256, 0, stream>>>(m2, NEG_INF, N_NODES);
    fill_kernel<<<(N_NODES + 255) / 256, 256, 0, stream>>>(den2, 0.f, N_NODES);
    fill_kernel<<<(N_NODES * NCLSV + 255) / 256, 256, 0, stream>>>(out, 0.f, N_NODES * NCLSV);

    // layer 1
    gemm1_kernel<<<dim3((ROW_TILES + 3) / 4, 8), 128, 0, stream>>>(x, pW1, h);
    attn_logits1<<<(N_NODES * HEADS + 255) / 256, 256, 0, stream>>>(h, as1, ad1, als1, ald1);
    e1_max<<<(N_ET * HEADS + 255) / 256, 256, 0, stream>>>(ei, als1, ald1, m1);
    e1_sum<<<(N_ET * HEADS + 255) / 256, 256, 0, stream>>>(ei, als1, ald1, m1, den1);
    e1_msg<<<N_ET, 256, 0, stream>>>(ei, als1, ald1, m1, den1, h, out1);
    elu_bias<<<(N_NODES * HH + 255) / 256, 256, 0, stream>>>(out1, b1);

    // layer 2
    gemm2_kernel<<<(ROW_TILES + 3) / 4, 128, 0, stream>>>(out1, pW2, h2);
    attn_logits2<<<(N_NODES + 255) / 256, 256, 0, stream>>>(h2, as2, ad2, als2v, ald2v);
    e2_max<<<(N_ET + 255) / 256, 256, 0, stream>>>(ei, als2v, ald2v, m2);
    e2_sum<<<(N_ET + 255) / 256, 256, 0, stream>>>(ei, als2v, ald2v, m2, den2);
    e2_msg<<<(N_ET * NCLSV + 255) / 256, 256, 0, stream>>>(ei, als2v, ald2v, m2, den2, h2, out);
    bias2_kernel<<<(N_NODES * NCLSV + 255) / 256, 256, 0, stream>>>(out, b2);
}
